// STC_encoder_44160853738179
// MI455X (gfx1250) — matrix operand order repaired and checked by reading, not compiled, not run on hardware
//
#include <hip/hip_runtime.h>
#include <hip/hip_bf16.h>
#include <float.h>

// ---------------------------------------------------------------------------
// Problem constants (match reference)
// ---------------------------------------------------------------------------
#define BN   100000      // batch nodes
#define UN   50000       // unique neighbors
#define TOT1 (BN + UN)   // rows of h
#define FD   512         // feature dim
#define DD   128         // embedding dim
#define KN   10          // neighbors per node
#define ALPHA 0.2f
#define K3   (2 * FD + DD)   // 1152

typedef __attribute__((ext_vector_type(16))) __bf16 v16bf;
typedef __attribute__((ext_vector_type(8)))  __bf16 v8bf;
typedef __attribute__((ext_vector_type(4)))  __bf16 v4bf;
typedef __attribute__((ext_vector_type(2)))  __bf16 v2bf;
typedef __attribute__((ext_vector_type(8)))  float  v8f;
typedef __attribute__((ext_vector_type(8)))  unsigned int v8u;

#define MT   4           // M sub-tiles (16 rows each) per block
#define ROWS (MT * 16)   // 64 rows per block

// ---------------------------------------------------------------------------
// Kernel 0: pre-pack weights to K-pair-packed bf16.
// Wpk[kp*DD + n] = { bf16(W[2kp, n]), bf16(W[2kp+1, n]) }  (lo, hi)
// B-fragment dword layout used here (contiguous k, as in A): VGPR v of a 32x16 bf16 B frag holds
// K = 16*half + 2v (lo) and +1 (hi) for column n.
// ---------------------------------------------------------------------------
__global__ __launch_bounds__(256) void
pack_w(const float* __restrict__ W, unsigned int* __restrict__ Wpk, int kpairs)
{
    int idx = blockIdx.x * 256 + threadIdx.x;
    if (idx >= kpairs * DD) return;
    int kp = idx / DD, n = idx % DD;
    v2bf p;
    p.x = (__bf16)W[(size_t)(2 * kp) * DD + n];
    p.y = (__bf16)W[(size_t)(2 * kp + 1) * DD + n];
    Wpk[idx] = __builtin_bit_cast(unsigned int, p);
}

// ---------------------------------------------------------------------------
// Kernel 1: h = [feats_batch ; feats_neigh] @ W   -> TOT1 x DD (fp32)
// 64-row blocks, 8 waves = 8 N-slices, 4 M-subtiles per wave (B-frag reuse).
// ---------------------------------------------------------------------------
#define K1_LDS_STRIDE 520   // 512 + 8 pad bf16; 1040B row = 4 dwords mod 64 banks

__global__ __launch_bounds__(256) void
k1_gemm_h(const float* __restrict__ fb, const float* __restrict__ fn,
          const unsigned int* __restrict__ Wpk, float* __restrict__ h)
{
    __shared__ __align__(16) __bf16 sA[ROWS * K1_LDS_STRIDE];

    const int row0 = blockIdx.x * ROWS;
    const int tid  = threadIdx.x;

    // Stage 64 x 512 fp32 -> bf16 LDS at float4 granularity (ds_store_b64)
    for (int idx = tid; idx < ROWS * (FD / 4); idx += 256) {
        int r  = idx >> 7;          // / (FD/4)
        int c4 = idx & 127;
        int grow = row0 + r;
        if (grow > TOT1 - 1) grow = TOT1 - 1;      // clamp tail rows
        const float* src = (grow < BN) ? (fb + (size_t)grow * FD)
                                       : (fn + (size_t)(grow - BN) * FD);
        float4 v = *(const float4*)(src + 4 * c4);
        v4bf p; p.x = (__bf16)v.x; p.y = (__bf16)v.y;
                p.z = (__bf16)v.z; p.w = (__bf16)v.w;
        *(v4bf*)(&sA[r * K1_LDS_STRIDE + 4 * c4]) = p;
    }
    __syncthreads();

    const int wave = tid >> 5;          // N-slice: cols [16*wave, 16*wave+16)
    const int lane = tid & 31;
    const int half = lane >> 4;
    const int lm   = lane & 15;
    const int n    = wave * 16 + lm;

    v8f acc[MT] = {};
    for (int k0 = 0; k0 < FD; k0 += 32) {
        // B fragment: 8 coalesced b32 loads of pre-packed bf16 pairs
        const unsigned int* wp = Wpk + (size_t)(k0 / 2 + 8 * half) * DD + n;
        v8u bu;
#pragma unroll
        for (int v = 0; v < 8; ++v) bu[v] = wp[(size_t)v * DD];
        v16bf bfrag = __builtin_bit_cast(v16bf, bu);

#pragma unroll
        for (int s = 0; s < MT; ++s) {
            // A fragment (16-bit A 16x32 layout), 2x ds_load_b128
            const __bf16* ar = &sA[(s * 16 + lm) * K1_LDS_STRIDE + k0 + half * 16];
            v8bf lo = *(const v8bf*)(ar);
            v8bf hi = *(const v8bf*)(ar + 8);
            v16bf afrag = __builtin_shufflevector(lo, hi,
                0,1,2,3,4,5,6,7,8,9,10,11,12,13,14,15);
            acc[s] = __builtin_amdgcn_wmma_f32_16x16x32_bf16(
                false, afrag, false, bfrag, (short)0, acc[s], false, false);
        }
    }

    // C/D layout: VGPR r -> M = r + 8*half, N = lm
#pragma unroll
    for (int s = 0; s < MT; ++s) {
#pragma unroll
        for (int r = 0; r < 8; ++r) {
            int grow = row0 + s * 16 + r + 8 * half;
            if (grow < TOT1)
                h[(size_t)grow * DD + n] = acc[s][r];
        }
    }
}

// ---------------------------------------------------------------------------
// Kernel 2: attention weights + segment-sum aggregation. One wave per node.
// Edges of node i are contiguous [10i, 10i+10) -> no atomics needed.
// ---------------------------------------------------------------------------
__device__ __forceinline__ float wave_sum32(float p) {
#pragma unroll
    for (int off = 16; off > 0; off >>= 1)
        p += __shfl_xor(p, off, 32);
    return p;
}

__device__ __forceinline__ float ntn(float x) {
    if (isnan(x)) return 0.0f;
    if (isinf(x)) return x > 0.0f ? FLT_MAX : -FLT_MAX;
    return x;
}

__global__ __launch_bounds__(256) void
k2_attn_agg(const float* __restrict__ h, const float* __restrict__ avec,
            const int* __restrict__ edge_dst, float* __restrict__ hp)
{
    const int node = blockIdx.x * 8 + (threadIdx.x >> 5);
    if (node >= BN) return;
    const int lane = threadIdx.x & 31;
    const int c    = lane * 4;

    const float4 hi4  = *(const float4*)(h + (size_t)node * DD + c);
    const float4 asrc = *(const float4*)(avec + c);
    const float4 adst = *(const float4*)(avec + DD + c);

    float s_src = wave_sum32(hi4.x * asrc.x + hi4.y * asrc.y +
                             hi4.z * asrc.z + hi4.w * asrc.w);

    float4 acc = {0.f, 0.f, 0.f, 0.f};
    float  esum = 0.f;
    const int* ed = edge_dst + node * KN;
#pragma unroll
    for (int t = 0; t < KN; ++t) {
        int j = ed[t];
        const float4 hd = *(const float4*)(h + (size_t)(BN + j) * DD + c);
        float s = s_src + wave_sum32(hd.x * adst.x + hd.y * adst.y +
                                     hd.z * adst.z + hd.w * adst.w);
        float lk = s > 0.f ? s : ALPHA * s;
        float e  = __expf(-lk);                 // v_exp_f32
        esum  += e;
        acc.x += e * hd.x; acc.y += e * hd.y;
        acc.z += e * hd.z; acc.w += e * hd.w;
    }

    float4 o;
    o.x = ntn(ntn(acc.x) / esum);
    o.y = ntn(ntn(acc.y) / esum);
    o.z = ntn(ntn(acc.z) / esum);
    o.w = ntn(ntn(acc.w) / esum);
    *(float4*)(hp + (size_t)node * DD + c) = o;
}

// ---------------------------------------------------------------------------
// Kernel 3: out = relu([feats_batch | h_prime | neigh_feats] @ Wd), K = 1152
// ---------------------------------------------------------------------------
#define K3_LDS_STRIDE 1160   // 1152 + 8 pad; 2320B row = 4 dwords mod 64 banks

__global__ __launch_bounds__(256) void
k3_gemm_out(const float* __restrict__ fb, const float* __restrict__ hp,
            const float* __restrict__ nf, const unsigned int* __restrict__ Wdpk,
            float* __restrict__ out)
{
    __shared__ __align__(16) __bf16 sA[ROWS * K3_LDS_STRIDE];

    const int row0 = blockIdx.x * ROWS;
    const int tid  = threadIdx.x;

    // Stage 64 x 1152 fp32 (3 segments; boundaries 512/640 are float4-aligned)
    for (int idx = tid; idx < ROWS * (K3 / 4); idx += 256) {
        int r  = idx / (K3 / 4);
        int c4 = idx % (K3 / 4);
        int grow = row0 + r;
        if (grow > BN - 1) grow = BN - 1;          // clamp tail rows
        int col = 4 * c4;
        const float* src;
        if (col < FD)             src = fb + (size_t)grow * FD + col;
        else if (col < FD + DD)   src = hp + (size_t)grow * DD + (col - FD);
        else                      src = nf + (size_t)grow * FD + (col - FD - DD);
        float4 v = *(const float4*)src;
        v4bf p; p.x = (__bf16)v.x; p.y = (__bf16)v.y;
                p.z = (__bf16)v.z; p.w = (__bf16)v.w;
        *(v4bf*)(&sA[r * K3_LDS_STRIDE + col]) = p;
    }
    __syncthreads();

    const int wave = tid >> 5;
    const int lane = tid & 31;
    const int half = lane >> 4;
    const int lm   = lane & 15;
    const int n    = wave * 16 + lm;

    v8f acc[MT] = {};
    for (int k0 = 0; k0 < K3; k0 += 32) {
        const unsigned int* wp = Wdpk + (size_t)(k0 / 2 + 8 * half) * DD + n;
        v8u bu;
#pragma unroll
        for (int v = 0; v < 8; ++v) bu[v] = wp[(size_t)v * DD];
        v16bf bfrag = __builtin_bit_cast(v16bf, bu);

#pragma unroll
        for (int s = 0; s < MT; ++s) {
            const __bf16* ar = &sA[(s * 16 + lm) * K3_LDS_STRIDE + k0 + half * 16];
            v8bf lo = *(const v8bf*)(ar);
            v8bf hi = *(const v8bf*)(ar + 8);
            v16bf afrag = __builtin_shufflevector(lo, hi,
                0,1,2,3,4,5,6,7,8,9,10,11,12,13,14,15);
            acc[s] = __builtin_amdgcn_wmma_f32_16x16x32_bf16(
                false, afrag, false, bfrag, (short)0, acc[s], false, false);
        }
    }

#pragma unroll
    for (int s = 0; s < MT; ++s) {
#pragma unroll
        for (int r = 0; r < 8; ++r) {
            int grow = row0 + s * 16 + r + 8 * half;
            if (grow < BN) {
                float v = acc[s][r];
                out[(size_t)grow * DD + n] = v > 0.f ? v : 0.f;
            }
        }
    }
}

// ---------------------------------------------------------------------------
// Host-side launcher
// ---------------------------------------------------------------------------
extern "C" void kernel_launch(void* const* d_in, const int* in_sizes, int n_in,
                              void* d_out, int out_size, void* d_ws, size_t ws_size,
                              hipStream_t stream)
{
    const float* fb   = (const float*)d_in[0];   // feats_batch  [BN,FD]
    const float* fn   = (const float*)d_in[1];   // feats_neigh  [UN,FD]
    const float* nf   = (const float*)d_in[2];   // neigh_feats  [BN,FD]
    const float* W    = (const float*)d_in[3];   // [FD,DD]
    const float* avec = (const float*)d_in[4];   // [1,2*DD]
    const float* Wd   = (const float*)d_in[5];   // [K3,DD]
    // d_in[6] = edge_src (repeat(arange(BN),KN)) -- structure exploited, unused
    const int*  edge_dst = (const int*)d_in[7];  // [BN*KN]
    float* out = (float*)d_out;                  // [BN,DD]

    // Workspace carve
    float* h  = (float*)d_ws;                             // TOT1 x DD
    float* hp = h + (size_t)TOT1 * DD;                    // BN x DD
    unsigned int* Wpk  = (unsigned int*)(hp + (size_t)BN * DD);  // (FD/2) x DD
    unsigned int* Wdpk = Wpk + (size_t)(FD / 2) * DD;            // (K3/2) x DD

    // 0) pre-pack weights to pair-packed bf16 (tiny)
    pack_w<<<((FD / 2) * DD + 255) / 256, 256, 0, stream>>>(W,  Wpk,  FD / 2);
    pack_w<<<((K3 / 2) * DD + 255) / 256, 256, 0, stream>>>(Wd, Wdpk, K3 / 2);
    // 1) h = concat @ W            (64-row tiles, tail clamped/guarded)
    k1_gemm_h<<<(TOT1 + ROWS - 1) / ROWS, 256, 0, stream>>>(fb, fn, Wpk, h);
    // 2) attention + aggregation   (8 nodes per block, one wave per node)
    k2_attn_agg<<<BN / 8, 256, 0, stream>>>(h, avec, edge_dst, hp);
    // 3) out = relu(concat @ Wd)
    k3_gemm_out<<<(BN + ROWS - 1) / ROWS, 256, 0, stream>>>(fb, hp, nf, Wdpk, out);
}
